// M2STN_15118284882370
// MI455X (gfx1250) — matrix operand order repaired and checked by reading, not compiled, not run on hardware
//
#include <hip/hip_runtime.h>

// Sizes from the reference
#define VN   4096   // graph nodes V (also GEMM M and K)
#define LN   24
#define BB   4
#define CN   32
#define NCOL 3072   // B*C*L  (GEMM N)
#define COUT 64
#define CCAT 224    // 7*C

typedef __attribute__((ext_vector_type(16))) __bf16 v16bf;
typedef __attribute__((ext_vector_type(8)))  __bf16 v8bf;
typedef __attribute__((ext_vector_type(4)))  __bf16 v4bf;
typedef __attribute__((ext_vector_type(8)))  float  v8f;
typedef __attribute__((ext_vector_type(4)))  float  v4f;

// ---------------------------------------------------------------------------
// 1) Transpose + convert adjacency: At[w][v] = bf16(A[v][w])   (K-major for WMMA A)
// block (32,8), grid (V/32, V/32)
// ---------------------------------------------------------------------------
__global__ __launch_bounds__(256) void agcn_prep_At(const float* __restrict__ A,
                                                    __bf16* __restrict__ At) {
    __shared__ float tile[32][33];
    const int tx = threadIdx.x;          // 0..31
    const int ty = threadIdx.y;          // 0..7
    const int w0 = blockIdx.x * 32;
    const int v0 = blockIdx.y * 32;
#pragma unroll
    for (int r = 0; r < 32; r += 8)
        tile[ty + r][tx] = A[(size_t)(v0 + ty + r) * VN + (w0 + tx)];
    __syncthreads();
#pragma unroll
    for (int r = 0; r < 32; r += 8)
        At[(size_t)(w0 + ty + r) * VN + (v0 + tx)] = (__bf16)tile[tx][ty + r];
}

// ---------------------------------------------------------------------------
// 2) Repack x [B,C,V,L] -> h0[n][v] bf16 with n = (b*C+c)*L + l  (K-major B op)
// ---------------------------------------------------------------------------
__global__ __launch_bounds__(256) void agcn_prep_X(const float* __restrict__ x,
                                                   __bf16* __restrict__ h0) {
    const int idx = blockIdx.x * 256 + threadIdx.x;   // n*VN + v
    const int v  = idx & (VN - 1);
    const int n  = idx >> 12;
    const int bc = n / LN;
    const int l  = n - bc * LN;
    h0[idx] = (__bf16)x[((size_t)bc * VN + v) * LN + l];
}

// ---------------------------------------------------------------------------
// 3) Diffusion GEMM:  Ct[n][w] = sum_v At[w][v] * Bt[n][v]
//    At: [VN][VN] bf16 (K contiguous), Bt: [NCOL][VN] bf16 (K contiguous)
//    Block tile 128(M) x 128(N), BK=64. 8 waves as 2(M) x 4(N); each wave owns
//    a 64x32 output patch = 4x2 WMMA 16x16 tiles (16 wmma per K-iter).
// ---------------------------------------------------------------------------
__global__ __launch_bounds__(256) void agcn_gemm(const __bf16* __restrict__ At,
                                                 const __bf16* __restrict__ Bt,
                                                 __bf16* __restrict__ Ct) {
    __shared__ __bf16 As[128][72];   // +8 bf16 pad (144B row stride)
    __shared__ __bf16 Bs[128][72];

    const int t    = threadIdx.x;
    const int m0   = blockIdx.x * 128;
    const int n0   = blockIdx.y * 128;
    const int wid  = t >> 5;
    const int lane = t & 31;
    const int half = lane >> 4;
    const int l16  = lane & 15;
    const int wm   = (wid >> 2) * 64;  // wave M origin in tile (0,64)
    const int wn   = (wid & 3) * 32;   // wave N origin in tile (0..96)

    v8f acc[4][2];
#pragma unroll
    for (int i = 0; i < 4; ++i)
#pragma unroll
        for (int j = 0; j < 2; ++j)
#pragma unroll
            for (int e = 0; e < 8; ++e) acc[i][j][e] = 0.0f;

    // global staging: both tiles are 128 rows x 64 bf16 -> 64B per thread
    const int sr = t >> 1;            // 0..127 staging row
    const int sc = (t & 1) * 32;      // 32 bf16 per thread

    const __bf16* ga = At + (size_t)(m0 + sr) * VN + sc;
    const __bf16* gb = Bt + (size_t)(n0 + sr) * VN + sc;

    for (int k0 = 0; k0 < VN; k0 += 64) {
        v8bf a0 = *(const v8bf*)(ga + k0 + 0);
        v8bf a1 = *(const v8bf*)(ga + k0 + 8);
        v8bf a2 = *(const v8bf*)(ga + k0 + 16);
        v8bf a3 = *(const v8bf*)(ga + k0 + 24);
        v8bf b0 = *(const v8bf*)(gb + k0 + 0);
        v8bf b1 = *(const v8bf*)(gb + k0 + 8);
        v8bf b2 = *(const v8bf*)(gb + k0 + 16);
        v8bf b3 = *(const v8bf*)(gb + k0 + 24);

        // head start for next K tile (global_prefetch_b8)
        __builtin_prefetch(ga + k0 + 64, 0, 1);
        __builtin_prefetch(gb + k0 + 64, 0, 1);

        __syncthreads();
        *(v8bf*)(&As[sr][sc +  0]) = a0;
        *(v8bf*)(&As[sr][sc +  8]) = a1;
        *(v8bf*)(&As[sr][sc + 16]) = a2;
        *(v8bf*)(&As[sr][sc + 24]) = a3;
        *(v8bf*)(&Bs[sr][sc +  0]) = b0;
        *(v8bf*)(&Bs[sr][sc +  8]) = b1;
        *(v8bf*)(&Bs[sr][sc + 16]) = b2;
        *(v8bf*)(&Bs[sr][sc + 24]) = b3;
        __syncthreads();

#pragma unroll
        for (int kk = 0; kk < 64; kk += 32) {
            v16bf afrag[4], bfrag[2];
#pragma unroll
            for (int i = 0; i < 4; ++i) {
                // A 16x32 bf16 fragment: elems 0..7 -> K = half*8+e ; 8..15 -> K = 16+half*8+e
                const int m = wm + i * 16 + l16;
                v8bf lo = *(const v8bf*)(&As[m][kk + half * 8]);
                v8bf hi = *(const v8bf*)(&As[m][kk + 16 + half * 8]);
#pragma unroll
                for (int e = 0; e < 8; ++e) { afrag[i][e] = lo[e]; afrag[i][e + 8] = hi[e]; }
            }
#pragma unroll
            for (int j = 0; j < 2; ++j) {
                // B 32x16 bf16 fragment: elems 0..7 -> K = half*8+e ; 8..15 -> K = 16+half*8+e
                const int n = wn + j * 16 + l16;
                v8bf bl = *(const v8bf*)(&Bs[n][kk + half * 8]);
                v8bf bh = *(const v8bf*)(&Bs[n][kk + 16 + half * 8]);
#pragma unroll
                for (int e = 0; e < 8; ++e) { bfrag[j][e] = bl[e]; bfrag[j][e + 8] = bh[e]; }
            }
#pragma unroll
            for (int i = 0; i < 4; ++i)
#pragma unroll
                for (int j = 0; j < 2; ++j)
                    acc[i][j] = __builtin_amdgcn_wmma_f32_16x16x32_bf16(
                        false, afrag[i], false, bfrag[j],
                        (short)0, acc[i][j], false, false);
        }
    }

    // Store transposed: Ct[n][w]; per lane 8 M-contiguous floats -> one 16B bf16 store
#pragma unroll
    for (int i = 0; i < 4; ++i)
#pragma unroll
        for (int j = 0; j < 2; ++j) {
            v8bf ov;
#pragma unroll
            for (int e = 0; e < 8; ++e) ov[e] = (__bf16)acc[i][j][e];
            const int w = m0 + wm + i * 16 + half * 8;
            const int n = n0 + wn + j * 16 + l16;
            *(v8bf*)(Ct + (size_t)n * VN + w) = ov;
        }
}

// ---------------------------------------------------------------------------
// 4) Channel mix: out[b,o,v,l] = bias[o] + sum_{g,c} W[o, g*32+c] * h_g[(b*32+c)*24+l][v]
//    One block per (b, l, 64-wide v tile).
// ---------------------------------------------------------------------------
__global__ __launch_bounds__(256) void agcn_mix(const __bf16* __restrict__ hs,
                                                const float* __restrict__ W,
                                                const float* __restrict__ bias,
                                                float* __restrict__ out) {
    __shared__ float lh[CCAT][68];    // 224 x 64 activations (padded)
    const int t  = threadIdx.x;
    const int vb = blockIdx.x & 63;
    const int l  = (blockIdx.x >> 6) % LN;
    const int b  = blockIdx.x / (64 * LN);
    const int v0 = vb * 64;

    const int rr0 = t >> 4;           // 0..15
    const int c4  = (t & 15) * 4;     // 4 bf16 per thread per row
#pragma unroll
    for (int it = 0; it < 14; ++it) {
        const int rr = rr0 + it * 16;          // 0..223
        const int g  = rr >> 5;
        const int c  = rr & 31;
        const __bf16* src = hs + (size_t)g * ((size_t)NCOL * VN)
                               + (size_t)((b * CN + c) * LN + l) * VN + v0 + c4;
        v4bf d = *(const v4bf*)src;
        v4f f;
#pragma unroll
        for (int e = 0; e < 4; ++e) f[e] = (float)d[e];
        *(v4f*)(&lh[rr][c4]) = f;
    }
    __syncthreads();

    const int v  = t & 63;
    const int og = t >> 6;            // 0..3
#pragma unroll 4
    for (int j = 0; j < 16; ++j) {
        const int o = og * 16 + j;
        float acc = bias[o];
        const float* wrow = W + o * CCAT;
#pragma unroll 8
        for (int cc = 0; cc < CCAT; ++cc)
            acc = fmaf(wrow[cc], lh[cc][v], acc);
        out[((size_t)(b * COUT + o) * VN + v0 + v) * LN + l] = acc;
    }
}

// ---------------------------------------------------------------------------
extern "C" void kernel_launch(void* const* d_in, const int* in_sizes, int n_in,
                              void* d_out, int out_size, void* d_ws, size_t ws_size,
                              hipStream_t stream) {
    (void)in_sizes; (void)n_in; (void)out_size; (void)ws_size;

    const float* x    = (const float*)d_in[0];
    const float* a[3] = { (const float*)d_in[1], (const float*)d_in[2], (const float*)d_in[3] };
    const float* W    = (const float*)d_in[4];
    const float* bias = (const float*)d_in[5];
    float* out        = (float*)d_out;

    const size_t VV = (size_t)VN * VN;        // 16.78M
    const size_t NV = (size_t)NCOL * VN;      // 12.58M
    __bf16* At = (__bf16*)d_ws;               // 3 * VV bf16
    __bf16* hs = At + 3 * VV;                 // 7 * NV bf16 (h0..h6)

    // Stage operands in bf16, K-major
    dim3 tb(32, 8);
    dim3 tg(VN / 32, VN / 32);
    for (int s = 0; s < 3; ++s)
        agcn_prep_At<<<tg, tb, 0, stream>>>(a[s], At + s * VV);
    agcn_prep_X<<<(unsigned)(NV / 256), 256, 0, stream>>>(x, hs);  // h0

    // 6 diffusion GEMMs: h1=A0ᵀh0, h2=A0ᵀh1, h3=A1ᵀh0, h4=A1ᵀh3, h5=A2ᵀh0, h6=A2ᵀh5
    dim3 gg(VN / 128, NCOL / 128);
    agcn_gemm<<<gg, 256, 0, stream>>>(At + 0 * VV, hs + 0 * NV, hs + 1 * NV);
    agcn_gemm<<<gg, 256, 0, stream>>>(At + 0 * VV, hs + 1 * NV, hs + 2 * NV);
    agcn_gemm<<<gg, 256, 0, stream>>>(At + 1 * VV, hs + 0 * NV, hs + 3 * NV);
    agcn_gemm<<<gg, 256, 0, stream>>>(At + 1 * VV, hs + 3 * NV, hs + 4 * NV);
    agcn_gemm<<<gg, 256, 0, stream>>>(At + 2 * VV, hs + 0 * NV, hs + 5 * NV);
    agcn_gemm<<<gg, 256, 0, stream>>>(At + 2 * VV, hs + 5 * NV, hs + 6 * NV);

    // Final 1x1-conv channel mix
    agcn_mix<<<BB * LN * (VN / 64), 256, 0, stream>>>(hs, W, bias, out);
}
